// TwoBranchDH_SFNN_86474871538260
// MI455X (gfx1250) — compile-verified, matrix operand order checked by reading, not run on hardware
//
#include <hip/hip_runtime.h>

// ---------------------------------------------------------------------------
// DH-SNN two-branch forward for MI455X (gfx1250, wave32, WMMA).
//   0a. input fp32 -> bf16 (staged in d_ws; 102 MB, L2-resident for phase 1)
//   0b. W1|W2 -> Wbf [1024,512] bf16, Wr -> Wrbf [16,512] bf16 (zero-pad)
//   1.  branch GEMM: d1,d2 = Xbf x Wbf^T (+bias). bf16 WMMA, f32 acc,
//       results rounded to bf16 (consistent with bf16 operand quantization).
//       B tile staged once per block in LDS; A and B fragments both
//       software-pipelined (partial loadcnt/dscnt waits, no full stalls).
//   2.  per-(b,h) scan over T=784, 2 h-chains per thread, bf16 in / bf16 out,
//       recurrence in fp32 (spikes reuse the Xbf region; Xbf dead after ph.1)
//   3.  readout GEMM via WMMA, O padded 10 -> 16, Wr tile in LDS
// ---------------------------------------------------------------------------

#define BATCH   128
#define TSTEPS  784
#define IDIM    512
#define HN      512
#define ODIM    10
#define MROWS   (BATCH * TSTEPS)      // 100352
#define KDIM    IDIM
#define BSTRIDE 520                   // 512 + 8 ushorts pad: LDS bank shift 4/row

typedef __attribute__((ext_vector_type(16))) __bf16 v16bf;
typedef __attribute__((ext_vector_type(8)))  float  v8f;

union FragU { uint4 u[2]; v16bf v; };

__device__ __forceinline__ unsigned short f2bf(float x) {
    unsigned int u = __float_as_uint(x);
    u += 0x7FFFu + ((u >> 16) & 1u);      // round-to-nearest-even
    return (unsigned short)(u >> 16);
}

__device__ __forceinline__ float sigm(float x) {
    return 1.0f / (1.0f + __expf(-x));
}

// One 16x32 bf16 WMMA fragment for this lane (ISA 7.12.2 layout):
// lane<16 -> K {0..7,16..23}, lane>=16 -> K {8..15,24..31}.
template <typename P>
__device__ __forceinline__ v16bf load_frag_bf(P rowp, int hi) {
    FragU f;
    f.u[0] = *reinterpret_cast<const uint4*>(rowp + (hi << 3));
    f.u[1] = *reinterpret_cast<const uint4*>(rowp + 16 + (hi << 3));
    return f.v;
}

// ---------------------------------------------------------------- phase 0a
__global__ __launch_bounds__(256) void cvt_input_kernel(
    const float* __restrict__ in, unsigned short* __restrict__ out)
{
    size_t i = ((size_t)blockIdx.x * 256 + threadIdx.x) * 4;
    float4 v = *reinterpret_cast<const float4*>(in + i);
    ushort4 o;
    o.x = f2bf(v.x); o.y = f2bf(v.y); o.z = f2bf(v.z); o.w = f2bf(v.w);
    *reinterpret_cast<ushort4*>(out + i) = o;
}

// ---------------------------------------------------------------- phase 0b
__global__ __launch_bounds__(256) void cvt_weights_kernel(
    const float* __restrict__ W1, const float* __restrict__ W2,
    const float* __restrict__ Wr,
    unsigned short* __restrict__ Wbf, unsigned short* __restrict__ Wrbf)
{
    int idx = blockIdx.x * 256 + threadIdx.x;
    const int NW = 1024 * 512;
    if (idx < NW) {
        float v = (idx < 512 * 512) ? W1[idx] : W2[idx - 512 * 512];
        Wbf[idx] = f2bf(v);
    } else {
        int j = idx - NW;                 // 0 .. 16*512-1
        int row = j >> 9, col = j & 511;
        Wrbf[j] = (row < ODIM) ? f2bf(Wr[row * 512 + col]) : (unsigned short)0;
    }
}

// ---------------------------------------------------------------- phase 1
// Block: 256 thr = 8 waves. Block tile 256M x 64N; wave tile 32M x 64N.
// B tile k 0..255 only (64 x 256 bf16 = 32KB) staged once in LDS; no barriers in K loop.
// N in [0,1024): [0,512) -> d1 (+b1), [512,1024) -> d2 (+b2). d* are bf16.
__global__ __launch_bounds__(256) void gemm_branch_kernel(
    const unsigned short* __restrict__ Abf,    // [MROWS,512] bf16
    const unsigned short* __restrict__ Wbf,    // [1024,512]  bf16
    const float* __restrict__ b1, const float* __restrict__ b2,
    unsigned short* __restrict__ d1, unsigned short* __restrict__ d2)
{
    __shared__ unsigned short ldsB[64 * BSTRIDE];   // 66,560 B

    const int tid    = threadIdx.x;
    const int lane   = tid & 31;
    const int laneLo = lane & 15;
    const int hi     = lane >> 4;
    const int wave   = tid >> 5;
    const int MR = blockIdx.x * 256 + wave * 32;
    const int NR = blockIdx.y * 64;

    // Cooperative, coalesced stage of B tile k 0..255 only (half of K).
    #pragma unroll
    for (int i = tid; i < 64 * 32; i += 256) {       // 16B chunks
        int row = i >> 5, chunk = (i & 31) << 3;     // chunk in ushorts
        uint4 v = *reinterpret_cast<const uint4*>(
            Wbf + (size_t)(NR + row) * KDIM + chunk);
        *reinterpret_cast<uint4*>(&ldsB[row * BSTRIDE + chunk]) = v;
    }
    __syncthreads();

    v8f acc[8];
    #pragma unroll
    for (int i = 0; i < 8; ++i) acc[i] = v8f{};

    const unsigned short* arow0 = Abf + (size_t)(MR + laneLo) * KDIM;
    const unsigned short* arow1 = arow0 + (size_t)16 * KDIM;
    const unsigned short* lb    = &ldsB[laneLo * BSTRIDE];

    // Software pipeline: A fragments (global) and B fragments (LDS) both
    // prefetched one stage ahead -> partial loadcnt/dscnt waits only.
    v16bf a0 = load_frag_bf(arow0, hi);
    v16bf a1 = load_frag_bf(arow1, hi);
    v16bf bcur = load_frag_bf(lb, hi);               // (nt=0, kb=0)

    #pragma unroll 4
    for (int kb = 0; kb < KDIM; kb += 32) {
        const bool more = (kb + 32) < KDIM;
        v16bf a0n, a1n;
        if (more) {
            a0n = load_frag_bf(arow0 + kb + 32, hi);
            a1n = load_frag_bf(arow1 + kb + 32, hi);
        }
        #pragma unroll
        for (int nt = 0; nt < 4; ++nt) {
            // next B fragment: (nt+1, kb) or (0, kb+32); dummy reload at end
            int nnt = (nt + 1) & 3;
            int nkb = (nt == 3 && more) ? (kb + 32) : (nt == 3 ? kb : kb);
            v16bf bnxt = load_frag_bf(lb + (size_t)(nnt * 16) * BSTRIDE +
                                      ((nt == 3) ? nkb : kb), hi);
            acc[nt]     = __builtin_amdgcn_wmma_f32_16x16x32_bf16(
                              false, a0, false, bcur, (short)0, acc[nt], false, false);
            acc[4 + nt] = __builtin_amdgcn_wmma_f32_16x16x32_bf16(
                              false, a1, false, bcur, (short)0, acc[4 + nt], false, false);
            bcur = bnxt;
        }
        if (more) { a0 = a0n; a1 = a1n; }
    }

    // C/D layout: lane<16 -> M = base+r, lane>=16 -> M = base+r+8, N = NR+nt*16+laneLo
    #pragma unroll
    for (int nt = 0; nt < 4; ++nt) {
        int n = NR + nt * 16 + laneLo;               // uniform: NR tile of 64
        unsigned short* dst; float bias; int ncol;
        if (n < HN) { dst = d1; bias = b1[n];      ncol = n; }
        else        { dst = d2; bias = b2[n - HN]; ncol = n - HN; }
        #pragma unroll
        for (int i = 0; i < 2; ++i) {
            #pragma unroll
            for (int r = 0; r < 8; ++r) {
                int m = MR + i * 16 + r + hi * 8;
                dst[(size_t)m * HN + ncol] = f2bf(acc[i * 4 + nt][r] + bias);
            }
        }
    }
}

// ---------------------------------------------------------------- phase 2
// Two (b,h) chains per thread; one b32 load per branch per step (ushort2
// bf16 -> fp32 via shift/and), recurrence in fp32, coalesced in h.
__global__ __launch_bounds__(256) void scan_kernel(
    const unsigned short* __restrict__ d1a, const unsigned short* __restrict__ d2a,
    const float* __restrict__ tau_m, const float* __restrict__ tau_n1,
    const float* __restrict__ tau_n2,
    const float* __restrict__ mem0, const float* __restrict__ spike0,
    unsigned short* __restrict__ spikes_bf)
{
    int t2 = blockIdx.x * 256 + threadIdx.x;     // 0 .. B*H/2-1
    int b  = t2 >> 8;                            // 256 h-pairs per batch entry
    int hp = (t2 & 255) * 2;

    float a0 = sigm(tau_m[hp]),  a1 = sigm(tau_m[hp + 1]);
    float p0 = sigm(tau_n1[hp]), p1 = sigm(tau_n1[hp + 1]);
    float q0 = sigm(tau_n2[hp]), q1 = sigm(tau_n2[hp + 1]);
    float oa0 = 1.0f - a0, oa1 = 1.0f - a1;
    float op0 = 1.0f - p0, op1 = 1.0f - p1;
    float oq0 = 1.0f - q0, oq1 = 1.0f - q1;

    int sidx = b * HN + hp;
    float2 mem = *reinterpret_cast<const float2*>(mem0 + sidx);
    float2 spk = *reinterpret_cast<const float2*>(spike0 + sidx);
    float dd10 = 0.0f, dd11 = 0.0f, dd20 = 0.0f, dd21 = 0.0f;
    size_t base = (size_t)b * TSTEPS * HN + hp;

    #pragma unroll 4
    for (int t = 0; t < TSTEPS; ++t) {
        size_t off = base + (size_t)t * HN;
        unsigned int u1 = *reinterpret_cast<const unsigned int*>(d1a + off);
        unsigned int u2 = *reinterpret_cast<const unsigned int*>(d2a + off);
        float x10 = __uint_as_float(u1 << 16);
        float x11 = __uint_as_float(u1 & 0xFFFF0000u);
        float x20 = __uint_as_float(u2 << 16);
        float x21 = __uint_as_float(u2 & 0xFFFF0000u);
        dd10 = p0 * dd10 + op0 * x10;  dd11 = p1 * dd11 + op1 * x11;
        dd20 = q0 * dd20 + oq0 * x20;  dd21 = q1 * dd21 + oq1 * x21;
        mem.x = mem.x * a0 + oa0 * (dd10 + dd20) - spk.x;   // V_TH = 1
        mem.y = mem.y * a1 + oa1 * (dd11 + dd21) - spk.y;
        spk.x = (mem.x - 1.0f > 0.0f) ? 1.0f : 0.0f;
        spk.y = (mem.y - 1.0f > 0.0f) ? 1.0f : 0.0f;
        unsigned int s = (spk.x > 0.0f ? 0x3F80u : 0u) |
                         (spk.y > 0.0f ? 0x3F800000u : 0u);
        *reinterpret_cast<unsigned int*>(spikes_bf + off) = s;
    }
}

// ---------------------------------------------------------------- phase 3
// Wave tile 16M x 16N (N padded to 16, store only 10). 8 waves / block.
// Wr tile staged in LDS (shared by all 8 waves); A and B pipelined.
__global__ __launch_bounds__(256) void readout_kernel(
    const unsigned short* __restrict__ spikes_bf,   // [MROWS,512] bf16
    const unsigned short* __restrict__ Wrbf,        // [16,512] bf16, rows>=10 zero
    const float* __restrict__ br, float* __restrict__ out)
{
    __shared__ unsigned short ldsW[16 * BSTRIDE];   // 16,640 B

    const int tid    = threadIdx.x;
    const int lane   = tid & 31;
    const int laneLo = lane & 15;
    const int hi     = lane >> 4;
    const int wave   = tid >> 5;
    const int MR = (blockIdx.x * 8 + wave) * 16;

    #pragma unroll
    for (int i = tid; i < 16 * 32; i += 256) {
        int row = i >> 5, chunk = (i & 31) << 3;
        uint4 v = *reinterpret_cast<const uint4*>(Wrbf + (size_t)row * HN + chunk);
        *reinterpret_cast<uint4*>(&ldsW[row * BSTRIDE + chunk]) = v;
    }
    __syncthreads();

    v8f acc = v8f{};
    const unsigned short* arow = spikes_bf + (size_t)(MR + laneLo) * HN;
    const unsigned short* lb   = &ldsW[laneLo * BSTRIDE];

    v16bf a_cur = load_frag_bf(arow, hi);
    v16bf b_cur = load_frag_bf(lb, hi);
    #pragma unroll 4
    for (int kb = 0; kb < HN; kb += 32) {
        const bool more = (kb + 32) < HN;
        v16bf a_nxt, b_nxt;
        if (more) {
            a_nxt = load_frag_bf(arow + kb + 32, hi);
            b_nxt = load_frag_bf(lb + kb + 32, hi);
        }
        acc = __builtin_amdgcn_wmma_f32_16x16x32_bf16(
                  false, a_cur, false, b_cur, (short)0, acc, false, false);
        if (more) { a_cur = a_nxt; b_cur = b_nxt; }
    }

    int n = laneLo;
    if (n < ODIM) {
        float bias = br[n];
        #pragma unroll
        for (int r = 0; r < 8; ++r) {
            int m = MR + r + hi * 8;
            out[(size_t)m * ODIM + n] = acc[r] + bias;
        }
    }
}

// ---------------------------------------------------------------------------
extern "C" void kernel_launch(void* const* d_in, const int* in_sizes, int n_in,
                              void* d_out, int out_size, void* d_ws, size_t ws_size,
                              hipStream_t stream)
{
    const float* input  = (const float*)d_in[0];
    const float* W1     = (const float*)d_in[1];
    const float* b1     = (const float*)d_in[2];
    const float* W2     = (const float*)d_in[3];
    const float* b2     = (const float*)d_in[4];
    const float* tau_m  = (const float*)d_in[5];
    const float* tau_n1 = (const float*)d_in[6];
    const float* tau_n2 = (const float*)d_in[7];
    const float* Wr     = (const float*)d_in[8];
    const float* br     = (const float*)d_in[9];
    const float* mem0   = (const float*)d_in[10];
    const float* spike0 = (const float*)d_in[11];
    float* out = (float*)d_out;

    char* ws = (char*)d_ws;
    const size_t szA  = (size_t)MROWS * KDIM * sizeof(unsigned short); // 102.8 MB
    const size_t szD  = (size_t)MROWS * HN * sizeof(unsigned short);   // 102.8 MB
    unsigned short* Abf  = (unsigned short*)(ws);
    unsigned short* d1   = (unsigned short*)(ws + szA);
    unsigned short* d2   = (unsigned short*)(ws + szA + szD);
    unsigned short* Wbf  = (unsigned short*)(ws + szA + 2 * szD);
    unsigned short* Wrbf = Wbf + (size_t)1024 * 512;
    unsigned short* spikes_bf = Abf;   // reuse: Abf dead after phase 1

    // Phase 0a: input fp32 -> bf16 (4 elems/thread)
    {
        int nblk = (int)(((size_t)MROWS * KDIM / 4) / 256);   // 50176, exact
        cvt_input_kernel<<<nblk, 256, 0, stream>>>(input, Abf);
    }
    // Phase 0b: weights -> bf16 (+ Wr zero-pad to 16 rows)
    {
        int total = 1024 * 512 + 16 * 512;
        cvt_weights_kernel<<<(total + 255) / 256, 256, 0, stream>>>(W1, W2, Wr, Wbf, Wrbf);
    }
    // Phase 1: branch GEMMs (WMMA bf16, B in LDS, pipelined)
    {
        dim3 grid(MROWS / 256, 1024 / 64, 1);                 // 392 x 16
        gemm_branch_kernel<<<grid, 256, 0, stream>>>(Abf, Wbf, b1, b2, d1, d2);
    }
    // Phase 2: temporal scan (2 chains/thread)
    {
        int nblk = (BATCH * HN / 2) / 256;                    // 128
        scan_kernel<<<nblk, 256, 0, stream>>>(d1, d2, tau_m, tau_n1, tau_n2,
                                              mem0, spike0, spikes_bf);
    }
    // Phase 3: readout (WMMA bf16, O padded to 16)
    {
        int nblk = (MROWS / 16) / 8;                          // 784
        readout_kernel<<<nblk, 256, 0, stream>>>(spikes_bf, Wrbf, br, out);
    }
}
